// Chord_Note_Conv_61177514165036
// MI455X (gfx1250) — matrix operand order repaired and checked by reading, not compiled, not run on hardware
//
#include <hip/hip_runtime.h>

// ---------------------------------------------------------------------------
// Shapes
// ---------------------------------------------------------------------------
#define B_    16
#define S_    2048
#define SP_   2050          // padded sequence (+1 zero row on each side)
#define NG_   32            // groups of 4 position tiles (each wave: N=64)
#define DIN_  512           // conv1 input channels (256 chord + 256 note)
#define C1_   256           // conv1 out channels
#define C2_   64            // conv2 out channels
#define V_    150           // vocab
#define VP_   160           // padded vocab (10 m-tiles)
#define KC1_  16            // 512/32 K-chunks
#define KC2_  8             // 256/32 K-chunks

typedef __attribute__((ext_vector_type(16))) _Float16 v16h;
typedef __attribute__((ext_vector_type(8)))  _Float16 v8h;
typedef __attribute__((ext_vector_type(8)))  float    v8f;
typedef __attribute__((ext_vector_type(2)))  float    v2f;

// ---------------------------------------------------------------------------
// K0: pack f32 weights into WMMA A-operand layout (f16), plus padded fc bias.
// A element mapping (16-bit A 16x32, wave32), this code's k order (same as B loads):
//   m = mt*16 + (lane&15)
//   k = kc*32 + 16*(lane>>4) + e
// pA1[t][kc:16][mt:16][lane:32][e:16]   (conv1, W[256][512][3])
// pA2[t][kc:8 ][mt:4 ][lane:32][e:16]   (conv2, W[64][256][3])
// pAf[kc:2][mt:10][lane:32][e:16]       (fc^T,  W[64][150], zero-padded M=160)
// fcbp[160] = fc_b zero-padded
// ---------------------------------------------------------------------------
__global__ __launch_bounds__(256) void pack_weights(
    const float* __restrict__ w1, const float* __restrict__ w2,
    const float* __restrict__ fw, const float* __restrict__ fb,
    _Float16* __restrict__ pA1, _Float16* __restrict__ pA2,
    _Float16* __restrict__ pAf, float* __restrict__ fcbp) {
  int tid = blockIdx.x * 256 + threadIdx.x;
  const int n1 = 3 * KC1_ * 16 * 32 * 16;  // 393216
  const int n2 = 3 * KC2_ * 4 * 32 * 16;   // 49152
  const int nf = 2 * 10 * 32 * 16;         // 10240
  if (tid < n1) {
    int e = tid & 15, lane = (tid >> 4) & 31, mt = (tid >> 9) & 15,
        kc = (tid >> 13) & 15, t = tid >> 17;
    int m = mt * 16 + (lane & 15);
    int k = kc * 32 + ((lane >> 4) << 4) + e;
    pA1[tid] = (_Float16)w1[(m * 512 + k) * 3 + t];
  } else if (tid < n1 + n2) {
    int id = tid - n1;
    int e = id & 15, lane = (id >> 4) & 31, mt = (id >> 9) & 3,
        kc = (id >> 11) & 7, t = id >> 14;
    int m = mt * 16 + (lane & 15);
    int k = kc * 32 + ((lane >> 4) << 4) + e;
    pA2[id] = (_Float16)w2[(m * 256 + k) * 3 + t];
  } else if (tid < n1 + n2 + nf) {
    int id = tid - n1 - n2;
    int e = id & 15, lane = (id >> 4) & 31;
    int blk = id >> 9;                // kc*10 + mt
    int mt = blk % 10, kc = blk / 10;
    int v = mt * 16 + (lane & 15);
    int k = kc * 32 + ((lane >> 4) << 4) + e;
    pAf[id] = (v < V_) ? (_Float16)fw[k * V_ + v] : (_Float16)0.0f;
  } else if (tid < n1 + n2 + nf + VP_) {
    int v = tid - n1 - n2 - nf;
    fcbp[v] = (v < V_) ? fb[v] : 0.0f;
  }
}

// ---------------------------------------------------------------------------
// K-pad: zero the +-1 halo rows of x (512 ch) and y1 (256 ch).
// ---------------------------------------------------------------------------
__global__ __launch_bounds__(256) void zero_pads(_Float16* __restrict__ x,
                                                 _Float16* __restrict__ y1) {
  int tid = blockIdx.x * 256 + threadIdx.x;
  const int nx = B_ * 2 * DIN_;  // 16384
  const int ny = B_ * 2 * C1_;   // 8192
  if (tid < nx) {
    int col = tid & 511, r = tid >> 9, b = r >> 1, hiRow = r & 1;
    x[((size_t)b * SP_ + (hiRow ? (SP_ - 1) : 0)) * DIN_ + col] = (_Float16)0.0f;
  } else if (tid < nx + ny) {
    int id = tid - nx;
    int col = id & 255, r = id >> 8, b = r >> 1, hiRow = r & 1;
    y1[((size_t)b * SP_ + (hiRow ? (SP_ - 1) : 0)) * C1_ + col] = (_Float16)0.0f;
  }
}

// ---------------------------------------------------------------------------
// K1: embeddings. One wave per (b,s). Lane owns 8 of 256 dims.
// Writes x[b][s+1][0:256]=chord_emb, x[b][s+1][256:512]=masked-mean note emb.
// ---------------------------------------------------------------------------
__global__ __launch_bounds__(256) void embed_kernel(
    const float* __restrict__ chord_emb, const float* __restrict__ note_emb,
    const int* __restrict__ note, const int* __restrict__ chord,
    _Float16* __restrict__ x) {
  int gtid = blockIdx.x * 256 + threadIdx.x;
  int wave = gtid >> 5;            // 0 .. B_*S_-1
  int lane = gtid & 31;
  int b = wave >> 11;              // / 2048
  int s = wave & 2047;

  int cidx = chord[b * S_ + s];    // wave-uniform
  const float* cp = chord_emb + (size_t)cidx * 256 + lane * 8;
  float ce[8];
#pragma unroll
  for (int i = 0; i < 8; ++i) ce[i] = cp[i];

  const int* np = note + ((size_t)b * S_ + s) * 16;
  float ne[8] = {0.f, 0.f, 0.f, 0.f, 0.f, 0.f, 0.f, 0.f};
  int cnt = 0;
  bool stop = false;
#pragma unroll
  for (int j = 0; j < 16; ++j) {
    int idx = np[j];               // wave-uniform
    stop = stop || (idx == 0);
    if (!stop) {
      ++cnt;
      const float* ep = note_emb + (size_t)idx * 256 + lane * 8;
#pragma unroll
      for (int i = 0; i < 8; ++i) ne[i] += ep[i];
    }
  }
  float inv = 1.0f / (float)(cnt > 0 ? cnt : 1);

  size_t row = ((size_t)b * SP_ + s + 1) * DIN_;
  v8h hc, hn;
#pragma unroll
  for (int i = 0; i < 8; ++i) {
    hc[i] = (_Float16)ce[i];
    hn[i] = (_Float16)(ne[i] * inv);
  }
  *(v8h*)(x + row + lane * 8) = hc;
  *(v8h*)(x + row + 256 + lane * 8) = hn;
}

// ---------------------------------------------------------------------------
// K2/K3: conv1d(k=3, pad=1) + bias + ReLU as shifted WMMA GEMMs.
// Each wave computes 32(M: tiles mp and mp+NMT/2) x 64(N: 4 tiles).
// -> every B fragment feeds 2 WMMAs, every A fragment feeds 4 WMMAs:
//    ~1.5 b128 loads per WMMA in the dominant conv1 kernel.
// B-operand k order used here (matches pA): lane = n + 16*(k/16), e = k%16 -> contiguous when
// activations are channel-contiguous. C/D: m = r + 8*(lane/16), n = lane%16.
// ---------------------------------------------------------------------------
template <int CIN, int COUT, int NKC, bool PAD_OUT>
__global__ __launch_bounds__(256) void conv_wmma(
    const _Float16* __restrict__ xin, const _Float16* __restrict__ pA,
    const float* __restrict__ bias, _Float16* __restrict__ yout) {
  const int NMT = COUT / 16;
  const int HM  = NMT / 2;         // m-tile pair stride
  int wave = blockIdx.x * 8 + (threadIdx.x >> 5);
  int lane = threadIdx.x & 31;
  int mp = wave % HM;              // pair id: m-tiles (mp, mp+HM)
  int ng = (wave / HM) % NG_;      // group of 4 position tiles
  int b  = wave / (HM * NG_);
  int n  = lane & 15;
  int hi = lane >> 4;

  v8f acc0[4] = {{}, {}, {}, {}};
  v8f acc1[4] = {{}, {}, {}, {}};
#pragma unroll
  for (int t = 0; t < 3; ++t) {
    // tap t reads padded row (s0 + n) + t  (s0 = ng*64 + u*16, shift +1, tap -1)
    const _Float16* brow =
        xin + ((size_t)b * SP_ + ng * 64 + n + t) * CIN + hi * 16;
#pragma unroll
    for (int kc = 0; kc < NKC; ++kc) {
      const _Float16* abase =
          pA + ((size_t)(t * NKC + kc) * NMT * 32) * 16 + (size_t)lane * 16;
      v16h a0 = *(const v16h*)(abase + (size_t)mp * 512);
      v16h a1 = *(const v16h*)(abase + (size_t)(mp + HM) * 512);
#pragma unroll
      for (int u = 0; u < 4; ++u) {
        v16h bm = *(const v16h*)(brow + (size_t)u * 16 * CIN + kc * 32);
        acc0[u] = __builtin_amdgcn_wmma_f32_16x16x32_f16(
            false, a0, false, bm, (short)0, acc0[u], false, false);
        acc1[u] = __builtin_amdgcn_wmma_f32_16x16x32_f16(
            false, a1, false, bm, (short)0, acc1[u], false, false);
      }
    }
  }

  int mb0 = mp * 16 + hi * 8;
  int mb1 = (mp + HM) * 16 + hi * 8;
  v8f bv0 = *(const v8f*)(bias + mb0);    // 32B aligned, in-bounds
  v8f bv1 = *(const v8f*)(bias + mb1);
#pragma unroll
  for (int u = 0; u < 4; ++u) {
    v8h o0, o1;
#pragma unroll
    for (int r = 0; r < 8; ++r) {
      float v0 = acc0[u][r] + bv0[r];
      float v1 = acc1[u][r] + bv1[r];
      o0[r] = (_Float16)(v0 > 0.f ? v0 : 0.f);
      o1[r] = (_Float16)(v1 > 0.f ? v1 : 0.f);
    }
    size_t orow = PAD_OUT ? ((size_t)b * SP_ + 1 + ng * 64 + u * 16 + n)
                          : ((size_t)b * S_ + ng * 64 + u * 16 + n);
    *(v8h*)(yout + orow * COUT + mb0) = o0;
    *(v8h*)(yout + orow * COUT + mb1) = o1;
  }
}

// ---------------------------------------------------------------------------
// K4: fc 64 -> 150 (M padded to 160), f32 output + bias.
// Wave computes m-tile pair (mp, mp+5) x 64 positions. Only mt==9 (mp==4,
// second tile) is partial -> wave-uniform masked path.
// ---------------------------------------------------------------------------
__global__ __launch_bounds__(256) void fc_wmma(
    const _Float16* __restrict__ y2, const _Float16* __restrict__ pAf,
    const float* __restrict__ fcbp, float* __restrict__ out) {
  int wave = blockIdx.x * 8 + (threadIdx.x >> 5);
  int lane = threadIdx.x & 31;
  int mp = wave % 5;               // pair (mp, mp+5)
  int ng = (wave / 5) % NG_;
  int b  = wave / (5 * NG_);
  int n  = lane & 15;
  int hi = lane >> 4;

  v8f acc0[4] = {{}, {}, {}, {}};
  v8f acc1[4] = {{}, {}, {}, {}};
  const _Float16* brow = y2 + ((size_t)b * S_ + ng * 64 + n) * C2_ + hi * 16;
#pragma unroll
  for (int kc = 0; kc < 2; ++kc) {
    const _Float16* abase =
        pAf + ((size_t)kc * 10 * 32) * 16 + (size_t)lane * 16;
    v16h a0 = *(const v16h*)(abase + (size_t)mp * 512);
    v16h a1 = *(const v16h*)(abase + (size_t)(mp + 5) * 512);
#pragma unroll
    for (int u = 0; u < 4; ++u) {
      v16h bm = *(const v16h*)(brow + (size_t)u * 16 * C2_ + kc * 32);
      acc0[u] = __builtin_amdgcn_wmma_f32_16x16x32_f16(
          false, a0, false, bm, (short)0, acc0[u], false, false);
      acc1[u] = __builtin_amdgcn_wmma_f32_16x16x32_f16(
          false, a1, false, bm, (short)0, acc1[u], false, false);
    }
  }

  int vb0 = mp * 16 + hi * 8;
  int vb1 = (mp + 5) * 16 + hi * 8;
  v8f bv0 = *(const v8f*)(fcbp + vb0);    // padded bias, 32B aligned
  v8f bv1 = *(const v8f*)(fcbp + vb1);

#pragma unroll
  for (int u = 0; u < 4; ++u) {
    size_t orow = ((size_t)b * S_ + ng * 64 + u * 16 + n) * (size_t)V_;
    float* op = out + orow;
    // first tile (v in [mp*16, mp*16+16) subset) is always fully in-bounds
#pragma unroll
    for (int i = 0; i < 4; ++i) {
      v2f p;
      p[0] = acc0[u][2 * i] + bv0[2 * i];
      p[1] = acc0[u][2 * i + 1] + bv0[2 * i + 1];
      *(v2f*)(op + vb0 + 2 * i) = p;     // 8B aligned (150, vb0 even)
    }
    if (mp < 4) {
      // second tile fully in-bounds
#pragma unroll
      for (int i = 0; i < 4; ++i) {
        v2f p;
        p[0] = acc1[u][2 * i] + bv1[2 * i];
        p[1] = acc1[u][2 * i + 1] + bv1[2 * i + 1];
        *(v2f*)(op + vb1 + 2 * i) = p;
      }
    } else {
      // mt==9: v in [144,160) -> mask v<150 (wave-uniform branch)
#pragma unroll
      for (int r = 0; r < 8; ++r) {
        int v = vb1 + r;
        if (v < V_) op[v] = acc1[u][r] + bv1[r];
      }
    }
  }
}

// ---------------------------------------------------------------------------
// Launch
// ---------------------------------------------------------------------------
extern "C" void kernel_launch(void* const* d_in, const int* in_sizes, int n_in,
                              void* d_out, int out_size, void* d_ws,
                              size_t ws_size, hipStream_t stream) {
  const float* chord_emb = (const float*)d_in[0];
  const float* note_emb  = (const float*)d_in[1];
  const float* conv1_w   = (const float*)d_in[2];
  const float* conv1_b   = (const float*)d_in[3];
  const float* conv2_w   = (const float*)d_in[4];
  const float* conv2_b   = (const float*)d_in[5];
  const float* fc_w      = (const float*)d_in[6];
  const float* fc_b      = (const float*)d_in[7];
  const int*   note      = (const int*)d_in[8];
  const int*   chord     = (const int*)d_in[9];
  float* out = (float*)d_out;

  // workspace carve-up (256B aligned sections)
  char* ws = (char*)d_ws;
  size_t off = 0;
  auto carve = [&](size_t bytes) {
    char* p = ws + off;
    off = (off + bytes + 255) & ~(size_t)255;
    return p;
  };
  _Float16* x    = (_Float16*)carve((size_t)B_ * SP_ * DIN_ * 2);  // 33.6 MB
  _Float16* y1   = (_Float16*)carve((size_t)B_ * SP_ * C1_ * 2);   // 16.8 MB
  _Float16* y2   = (_Float16*)carve((size_t)B_ * S_  * C2_ * 2);   //  4.2 MB
  _Float16* pA1  = (_Float16*)carve((size_t)3 * KC1_ * 16 * 512 * 2);
  _Float16* pA2  = (_Float16*)carve((size_t)3 * KC2_ * 4 * 512 * 2);
  _Float16* pAf  = (_Float16*)carve((size_t)2 * 10 * 512 * 2);
  float*    fcbp = (float*)carve((size_t)VP_ * 4);
  (void)ws_size; (void)in_sizes; (void)n_in; (void)out_size;

  // K0: weight packing (+ padded fc bias)
  {
    int total = 3 * KC1_ * 16 * 512 + 3 * KC2_ * 4 * 512 + 2 * 10 * 512 + VP_;
    pack_weights<<<(total + 255) / 256, 256, 0, stream>>>(
        conv1_w, conv2_w, fc_w, fc_b, pA1, pA2, pAf, fcbp);
  }
  // K-pad: zero halos
  zero_pads<<<(B_ * 2 * (DIN_ + C1_) + 255) / 256, 256, 0, stream>>>(x, y1);

  // K1: embeddings — one wave per (b,s)
  embed_kernel<<<(B_ * S_ * 32) / 256, 256, 0, stream>>>(
      chord_emb, note_emb, note, chord, x);

  // K2: conv1 — waves = 8 m-pairs * 32 ng * 16 b = 4096, 8 waves/block
  conv_wmma<DIN_, C1_, KC1_, true><<<(8 * NG_ * B_) / 8, 256, 0, stream>>>(
      x, pA1, conv1_b, y1);

  // K3: conv2 — waves = 2 * 32 * 16 = 1024
  conv_wmma<C1_, C2_, KC2_, false><<<(2 * NG_ * B_) / 8, 256, 0, stream>>>(
      y1, pA2, conv2_b, y2);

  // K4: fc — waves = 5 * 32 * 16 = 2560
  fc_wmma<<<(5 * NG_ * B_) / 8, 256, 0, stream>>>(y2, pAf, fcbp, out);
}
